// Decoder_55585466745096
// MI455X (gfx1250) — compile-verified, matrix operand order checked by reading, not run on hardware
//
#include <hip/hip_runtime.h>
#include <cstdint>
#include <cstddef>

// ---------------------------------------------------------------------------
// Problem constants (B, L, H, IN, OUT, NH, HOR) = (128, 512, 512, 64, 64, 8, 24)
// ---------------------------------------------------------------------------
#define BB   128
#define LL   512
#define HH   512
#define INN  64
#define OUTN 64
#define NHD  8
#define HOR  24
#define HD   64
#define PRED_ELEMS (BB * HOR * OUTN)   // 196608 floats, then attention maps

typedef unsigned short bf16s;          // bf16 storage type (host-safe)
typedef __bf16 bf16t;
typedef __attribute__((ext_vector_type(16))) __bf16 v16bf;
typedef __attribute__((ext_vector_type(8)))  __bf16 v8bf;
typedef __attribute__((ext_vector_type(8)))  float  v8f;

#define DEV __device__ __forceinline__

// Native bf16 convert (lets clang emit v_cvt*_bf16_f32 instead of manual RNE)
DEV bf16s f2bf(float x) {
  return __builtin_bit_cast(bf16s, static_cast<bf16t>(x));
}
DEV float bf2f(bf16s s) { return __builtin_bit_cast(float, (uint32_t)s << 16); }
DEV void bfpair(uint32_t u, float &lo, float &hi) {
  lo = __builtin_bit_cast(float, u << 16);
  hi = __builtin_bit_cast(float, u & 0xFFFF0000u);
}
DEV float sigmoidf_(float x) { return 1.0f / (1.0f + __expf(-x)); }

// A/B fragment for V_WMMA_F32_16X16X32_BF16 from a row-major bf16 matrix.
// lane l: row (l&15); K elements: [8*half .. +7] and [16+8*half .. +7]
// -> VGPR0..3 then VGPR4..7, matching the ISA 16-bit 16x32 layout.
DEV v16bf load_frag_bf16(const bf16s* base, int ld, int kk, int lane) {
  int r = lane & 15, half = lane >> 4;
  const bf16s* p = base + (size_t)r * ld + kk + 8 * half;
  v8bf c0 = *reinterpret_cast<const v8bf*>(p);
  v8bf c1 = *reinterpret_cast<const v8bf*>(p + 16);
  return __builtin_shufflevector(c0, c1, 0,1,2,3,4,5,6,7,8,9,10,11,12,13,14,15);
}

// Same fragment but converting from an fp32 row-major source on the fly.
DEV v16bf load_frag_f32(const float* base, int ld, int kk, int lane) {
  int r = lane & 15, half = lane >> 4;
  const float* p = base + (size_t)r * ld + kk + 8 * half;
  v16bf f;
#pragma unroll
  for (int i = 0; i < 8; ++i) f[i]     = static_cast<bf16t>(p[i]);
#pragma unroll
  for (int i = 0; i < 8; ++i) f[8 + i] = static_cast<bf16t>(p[16 + i]);
  return f;
}

DEV v8f wmma_bf16(v16bf a, v16bf b, v8f c) {
  return __builtin_amdgcn_wmma_f32_16x16x32_bf16(false, a, false, b, (short)0, c,
                                                 false, false);
}

// ---------------------------------------------------------------------------
// Setup kernels
// ---------------------------------------------------------------------------
__global__ void cvt_f32_bf16_kernel(const float* __restrict__ src,
                                    bf16s* __restrict__ dst, int n) {
  int i = blockIdx.x * blockDim.x + threadIdx.x;
  if (i < n) dst[i] = f2bf(src[i]);
}

__global__ void init_state_kernel(const float* __restrict__ h0,
                                  const float* __restrict__ c0,
                                  const float* __restrict__ dec,
                                  float* __restrict__ c,
                                  bf16s* __restrict__ hbf,
                                  bf16s* __restrict__ xbf) {
  int i = blockIdx.x * blockDim.x + threadIdx.x;
  if (i < BB * HH) { c[i] = c0[i]; hbf[i] = f2bf(h0[i]); }
  if (i < BB * INN) xbf[i] = f2bf(dec[i]);
}

// ---------------------------------------------------------------------------
// KV projection:  P = enc(f32) @ W^T + bias  -> bf16 (B*L x H), z selects K/V
// block = 256 threads (8 waves); block covers 32 rows x 512 cols:
// each wave holds 2 M-tiles x 4 N-tiles (8 accumulators) so every B fragment
// is reused across two WMMAs. grid (2048, 2).
// ---------------------------------------------------------------------------
__global__ void kv_gemm_kernel(const float* __restrict__ enc,
                               const bf16s* __restrict__ Wkb,
                               const bf16s* __restrict__ Wvb,
                               const float* __restrict__ bk,
                               const float* __restrict__ bv,
                               bf16s* __restrict__ Kp,
                               bf16s* __restrict__ Vp) {
  int lane = threadIdx.x & 31, wid = threadIdx.x >> 5;
  size_t m0 = (size_t)blockIdx.x * 32;
  const bf16s* W   = blockIdx.y ? Wvb : Wkb;
  const float* bia = blockIdx.y ? bv  : bk;
  bf16s*       out = blockIdx.y ? Vp  : Kp;

  v8f acc[2][4];
#pragma unroll
  for (int mi = 0; mi < 2; ++mi)
#pragma unroll
    for (int j = 0; j < 4; ++j) acc[mi][j] = v8f{0,0,0,0,0,0,0,0};

  for (int kk = 0; kk < HH; kk += 32) {
    v16bf a0 = load_frag_f32(enc + m0 * HH,        HH, kk, lane);
    v16bf a1 = load_frag_f32(enc + (m0 + 16) * HH, HH, kk, lane);
#pragma unroll
    for (int j = 0; j < 4; ++j) {
      int n0 = (wid * 4 + j) * 16;
      v16bf b = load_frag_bf16(W + (size_t)n0 * HH, HH, kk, lane);
      acc[0][j] = wmma_bf16(a0, b, acc[0][j]);
      acc[1][j] = wmma_bf16(a1, b, acc[1][j]);
    }
  }
  int half = lane >> 4, nl = lane & 15;
#pragma unroll
  for (int mi = 0; mi < 2; ++mi)
#pragma unroll
    for (int j = 0; j < 4; ++j) {
      int n = (wid * 4 + j) * 16 + nl;
      float bvv = bia[n];
#pragma unroll
      for (int r = 0; r < 8; ++r) {
        size_t m = m0 + mi * 16 + r + 8 * half;
        out[m * HH + n] = f2bf(acc[mi][j][r] + bvv);
      }
    }
}

// ---------------------------------------------------------------------------
// Generic small GEMM:  out = A(bf16, M=128) @ W^T(bf16) + bias
// grid.x = 8 (16 rows each); wave covers TJ N-tiles; N = waves*TJ*16
// ---------------------------------------------------------------------------
__global__ void gemm_bf16_kernel(const bf16s* __restrict__ A, int lda,
                                 const bf16s* __restrict__ W, int ldw,
                                 const float* __restrict__ bias,
                                 bf16s* __restrict__ outb, int ldcb,
                                 float* __restrict__ outf, int ldcf,
                                 int K, int TJ) {
  int lane = threadIdx.x & 31, wid = threadIdx.x >> 5;
  int m0 = blockIdx.x * 16;
  const bf16s* Abase = A + (size_t)m0 * lda;

  v8f acc[4];
#pragma unroll
  for (int j = 0; j < 4; ++j) acc[j] = v8f{0,0,0,0,0,0,0,0};

  for (int kk = 0; kk < K; kk += 32) {
    v16bf a = load_frag_bf16(Abase, lda, kk, lane);
    for (int j = 0; j < TJ; ++j) {
      int n0 = (wid * TJ + j) * 16;
      v16bf b = load_frag_bf16(W + (size_t)n0 * ldw, ldw, kk, lane);
      acc[j] = wmma_bf16(a, b, acc[j]);
    }
  }
  int half = lane >> 4, nl = lane & 15;
  for (int j = 0; j < TJ; ++j) {
    int n = (wid * TJ + j) * 16 + nl;
    float bvv = bias ? bias[n] : 0.0f;
#pragma unroll
    for (int r = 0; r < 8; ++r) {
      int m = m0 + r + 8 * half;
      float v = acc[j][r] + bvv;
      if (outb) outb[(size_t)m * ldcb + n] = f2bf(v);
      if (outf) outf[(size_t)m * ldcf + n] = v;
    }
  }
}

// ---------------------------------------------------------------------------
// Fused gates GEMM + LSTM cell.
// gates = x@Wih^T + h@Whh^T + b ; accumulators map (m,n)->(batch,hidden),
// so i/f/g/o live in matching accumulators and the cell runs in registers.
// block = 512 threads (16 waves); wave covers 2 hidden tiles (32 cols) x 4 gates.
// ---------------------------------------------------------------------------
__global__ void gates_lstm_kernel(const bf16s* __restrict__ xbf,
                                  const bf16s* __restrict__ hin,
                                  const bf16s* __restrict__ Wih,
                                  const bf16s* __restrict__ Whh,
                                  const float* __restrict__ bih,
                                  const float* __restrict__ bhh,
                                  float* __restrict__ c,
                                  bf16s* __restrict__ hout) {
  int lane = threadIdx.x & 31, wid = threadIdx.x >> 5;
  int m0 = blockIdx.x * 16;

  v8f acc[2][4];
#pragma unroll
  for (int j = 0; j < 2; ++j)
#pragma unroll
    for (int g = 0; g < 4; ++g) acc[j][g] = v8f{0,0,0,0,0,0,0,0};

  // x @ W_ih^T  (K = 64)
  const bf16s* Ax = xbf + (size_t)m0 * INN;
#pragma unroll
  for (int kk = 0; kk < INN; kk += 32) {
    v16bf a = load_frag_bf16(Ax, INN, kk, lane);
#pragma unroll
    for (int j = 0; j < 2; ++j)
#pragma unroll
      for (int g = 0; g < 4; ++g) {
        int n0 = g * HH + (wid * 2 + j) * 16;
        v16bf b = load_frag_bf16(Wih + (size_t)n0 * INN, INN, kk, lane);
        acc[j][g] = wmma_bf16(a, b, acc[j][g]);
      }
  }
  // h @ W_hh^T  (K = 512)
  const bf16s* Ah = hin + (size_t)m0 * HH;
  for (int kk = 0; kk < HH; kk += 32) {
    v16bf a = load_frag_bf16(Ah, HH, kk, lane);
#pragma unroll
    for (int j = 0; j < 2; ++j)
#pragma unroll
      for (int g = 0; g < 4; ++g) {
        int n0 = g * HH + (wid * 2 + j) * 16;
        v16bf b = load_frag_bf16(Whh + (size_t)n0 * HH, HH, kk, lane);
        acc[j][g] = wmma_bf16(a, b, acc[j][g]);
      }
  }

  int half = lane >> 4, nl = lane & 15;
#pragma unroll
  for (int j = 0; j < 2; ++j) {
    int n = (wid * 2 + j) * 16 + nl;
    float Bi = bih[n]            + bhh[n];
    float Bf = bih[n + HH]       + bhh[n + HH];
    float Bg = bih[n + 2 * HH]   + bhh[n + 2 * HH];
    float Bo = bih[n + 3 * HH]   + bhh[n + 3 * HH];
#pragma unroll
    for (int r = 0; r < 8; ++r) {
      int m = m0 + r + 8 * half;
      size_t idx = (size_t)m * HH + n;
      float gi = acc[j][0][r] + Bi;
      float gf = acc[j][1][r] + Bf;
      float gg = acc[j][2][r] + Bg;
      float go = acc[j][3][r] + Bo;
      float cn = sigmoidf_(gf) * c[idx] + sigmoidf_(gi) * tanhf(gg);
      float hn = sigmoidf_(go) * tanhf(cn);
      c[idx]    = cn;
      hout[idx] = f2bf(hn);
    }
  }
}

// ---------------------------------------------------------------------------
// Attention: one wave per (batch, head). scores = scale * q.K, softmax over
// L=512, ctx = attn.V. K/V are bf16 (L2-resident: 134 MB < 192 MB L2).
// Writes attn map (f32) directly to d_out. block = 256 (8 waves), grid = 128.
// ---------------------------------------------------------------------------
__global__ void attention_kernel(const bf16s* __restrict__ qbf,
                                 const bf16s* __restrict__ Kp,
                                 const bf16s* __restrict__ Vp,
                                 bf16s* __restrict__ ctx,
                                 float* __restrict__ attn_out, int t) {
  __shared__ float lq[8][HD];
  __shared__ float lat[8][LL];
  int lane = threadIdx.x & 31, wid = threadIdx.x >> 5;
  int gw = blockIdx.x * 8 + wid;          // 0..1023
  int b = gw >> 3, hh = gw & 7;

  { // broadcast q for this (b, head) into LDS
    uint32_t u = *reinterpret_cast<const uint32_t*>(qbf + (size_t)b * HH + hh * HD + lane * 2);
    float lo, hi; bfpair(u, lo, hi);
    lq[wid][lane * 2] = lo; lq[wid][lane * 2 + 1] = hi;
  }
  __syncthreads();

  const bf16s* Kb = Kp + ((size_t)b * LL) * HH + hh * HD;
  float sc[16];
#pragma unroll
  for (int jj = 0; jj < 16; ++jj) {
    int l = jj * 32 + lane;
    const bf16s* row = Kb + (size_t)l * HH;
    float acc = 0.f;
#pragma unroll
    for (int k = 0; k < HD; k += 8) {
      uint4 u = *reinterpret_cast<const uint4*>(row + k);
      float a0,a1,a2,a3,a4,a5,a6,a7;
      bfpair(u.x,a0,a1); bfpair(u.y,a2,a3); bfpair(u.z,a4,a5); bfpair(u.w,a6,a7);
      acc += a0*lq[wid][k]   + a1*lq[wid][k+1] + a2*lq[wid][k+2] + a3*lq[wid][k+3]
           + a4*lq[wid][k+4] + a5*lq[wid][k+5] + a6*lq[wid][k+6] + a7*lq[wid][k+7];
    }
    sc[jj] = acc * 0.125f;                 // 1/sqrt(64)
  }
  // softmax over 512 (16 per lane x wave32 reduction)
  float mx = sc[0];
#pragma unroll
  for (int jj = 1; jj < 16; ++jj) mx = fmaxf(mx, sc[jj]);
#pragma unroll
  for (int off = 16; off >= 1; off >>= 1) mx = fmaxf(mx, __shfl_xor(mx, off, 32));
  float sum = 0.f;
#pragma unroll
  for (int jj = 0; jj < 16; ++jj) { sc[jj] = __expf(sc[jj] - mx); sum += sc[jj]; }
#pragma unroll
  for (int off = 16; off >= 1; off >>= 1) sum += __shfl_xor(sum, off, 32);
  float inv = 1.0f / sum;

  float* aout = attn_out + ((size_t)gw * HOR + t) * LL;   // (B,NH,HOR,L)
#pragma unroll
  for (int jj = 0; jj < 16; ++jj) {
    int l = jj * 32 + lane;
    float a = sc[jj] * inv;
    lat[wid][l] = a;
    aout[l] = a;
  }
  __syncthreads();

  // ctx[d] = sum_l attn[l] * V[l][d]; lanes over d (2 per lane)
  const bf16s* Vb = Vp + ((size_t)b * LL) * HH + hh * HD + lane * 2;
  float c0 = 0.f, c1 = 0.f;
#pragma unroll 4
  for (int l = 0; l < LL; ++l) {
    float a = lat[wid][l];
    uint32_t u = *reinterpret_cast<const uint32_t*>(Vb + (size_t)l * HH);
    float v0, v1; bfpair(u, v0, v1);
    c0 += a * v0; c1 += a * v1;
  }
  bf16s* cp = ctx + (size_t)b * HH + hh * HD + lane * 2;
  cp[0] = f2bf(c0); cp[1] = f2bf(c1);
}

// ---------------------------------------------------------------------------
// Host driver
// ---------------------------------------------------------------------------
extern "C" void kernel_launch(void* const* d_in, const int* in_sizes, int n_in,
                              void* d_out, int out_size, void* d_ws, size_t ws_size,
                              hipStream_t stream) {
  (void)in_sizes; (void)n_in; (void)out_size; (void)ws_size;

  const float* enc  = (const float*)d_in[0];
  const float* h0   = (const float*)d_in[1];
  const float* c0   = (const float*)d_in[2];
  const float* dec  = (const float*)d_in[3];
  const float* W_ih = (const float*)d_in[4];
  const float* W_hh = (const float*)d_in[5];
  const float* b_ih = (const float*)d_in[6];
  const float* b_hh = (const float*)d_in[7];
  const float* Wq   = (const float*)d_in[8];
  const float* bq   = (const float*)d_in[9];
  const float* Wk   = (const float*)d_in[10];
  const float* bk   = (const float*)d_in[11];
  const float* Wv   = (const float*)d_in[12];
  const float* bv   = (const float*)d_in[13];
  const float* Wo   = (const float*)d_in[14];
  const float* bo   = (const float*)d_in[15];
  const float* Wfc  = (const float*)d_in[16];
  const float* bfc  = (const float*)d_in[17];
  float* outp = (float*)d_out;

  // ---- workspace carve-out (~140 MB total) ----
  char* ws = (char*)d_ws;
  size_t off = 0;
  auto carve = [&](size_t bytes) -> void* {
    void* p = ws + off;
    off = (off + bytes + 255) & ~(size_t)255;
    return p;
  };
  bf16s* Wih_bf = (bf16s*)carve((size_t)4 * HH * INN * 2);   // 2048x64
  bf16s* Whh_bf = (bf16s*)carve((size_t)4 * HH * HH  * 2);   // 2048x512
  bf16s* Wq_bf  = (bf16s*)carve((size_t)HH * HH * 2);
  bf16s* Wk_bf  = (bf16s*)carve((size_t)HH * HH * 2);
  bf16s* Wv_bf  = (bf16s*)carve((size_t)HH * HH * 2);
  bf16s* Wo_bf  = (bf16s*)carve((size_t)HH * HH * 2);
  bf16s* Wfc_bf = (bf16s*)carve((size_t)OUTN * HH * 2);
  bf16s* Kp     = (bf16s*)carve((size_t)BB * LL * HH * 2);   // 64 MB
  bf16s* Vp     = (bf16s*)carve((size_t)BB * LL * HH * 2);   // 64 MB
  float* c_ws   = (float*)carve((size_t)BB * HH * 4);
  bf16s* hA     = (bf16s*)carve((size_t)BB * HH * 2);
  bf16s* hB     = (bf16s*)carve((size_t)BB * HH * 2);
  bf16s* x_bf   = (bf16s*)carve((size_t)BB * INN * 2);
  bf16s* q_bf   = (bf16s*)carve((size_t)BB * HH * 2);
  bf16s* ctx_bf = (bf16s*)carve((size_t)BB * HH * 2);
  bf16s* out_bf = (bf16s*)carve((size_t)BB * HH * 2);

  // ---- one-time: weight conversion to bf16 ----
  auto cvt = [&](const float* s, bf16s* d, int n) {
    cvt_f32_bf16_kernel<<<(n + 255) / 256, 256, 0, stream>>>(s, d, n);
  };
  cvt(W_ih, Wih_bf, 4 * HH * INN);
  cvt(W_hh, Whh_bf, 4 * HH * HH);
  cvt(Wq,   Wq_bf,  HH * HH);
  cvt(Wk,   Wk_bf,  HH * HH);
  cvt(Wv,   Wv_bf,  HH * HH);
  cvt(Wo,   Wo_bf,  HH * HH);
  cvt(Wfc,  Wfc_bf, OUTN * HH);

  init_state_kernel<<<(BB * HH + 255) / 256, 256, 0, stream>>>(h0, c0, dec,
                                                               c_ws, hA, x_bf);

  // ---- K/V projections (WMMA bf16 GEMM, z = {K, V}) ----
  kv_gemm_kernel<<<dim3((BB * LL) / 32, 2), 256, 0, stream>>>(
      enc, Wk_bf, Wv_bf, bk, bv, Kp, Vp);

  // ---- 24 sequential decode steps ----
  for (int t = 0; t < HOR; ++t) {
    const bf16s* hin = (t & 1) ? hB : hA;
    bf16s*       hout = (t & 1) ? hA : hB;

    gates_lstm_kernel<<<BB / 16, 512, 0, stream>>>(x_bf, hin, Wih_bf, Whh_bf,
                                                   b_ih, b_hh, c_ws, hout);
    // q = h @ Wq^T + bq
    gemm_bf16_kernel<<<BB / 16, 256, 0, stream>>>(hout, HH, Wq_bf, HH, bq,
                                                  q_bf, HH, (float*)nullptr, 0,
                                                  HH, 4);
    attention_kernel<<<(BB * NHD) / 8, 256, 0, stream>>>(q_bf, Kp, Vp, ctx_bf,
                                                         outp + PRED_ELEMS, t);
    // out = ctx @ Wo^T + bo
    gemm_bf16_kernel<<<BB / 16, 256, 0, stream>>>(ctx_bf, HH, Wo_bf, HH, bo,
                                                  out_bf, HH, (float*)nullptr, 0,
                                                  HH, 4);
    // pred = out @ Wfc^T + bfc  -> d_out[(b,t,:)] (f32) and next x (bf16)
    gemm_bf16_kernel<<<BB / 16, 64, 0, stream>>>(out_bf, HH, Wfc_bf, HH, bfc,
                                                 x_bf, OUTN,
                                                 outp + (size_t)t * OUTN,
                                                 HOR * OUTN, HH, 2);
  }
}
